// HardGAO_25778393710965
// MI455X (gfx1250) — compile-verified, matrix operand order checked by reading, not run on hardware
//
#include <hip/hip_runtime.h>
#include <hip/hip_bf16.h>
#include <math.h>

typedef __attribute__((ext_vector_type(16))) _Float16 v16h;
typedef __attribute__((ext_vector_type(8)))  float    v8f;

#define FDIM 256
#define DNEI 64
#define TOPK 8
#define HEADS 8
#define ODIM 32
#define NEG_SLOPE 0.2f

// ---------------------------------------------------------------------------
// 1. ||p||
// ---------------------------------------------------------------------------
__global__ void pnorm_kernel(const float* __restrict__ p, float* __restrict__ pn) {
    __shared__ float s[FDIM];
    int t = threadIdx.x;
    float v = p[t];
    s[t] = v * v;
    __syncthreads();
    for (int st = FDIM / 2; st > 0; st >>= 1) {
        if (t < st) s[t] += s[t + st];
        __syncthreads();
    }
    if (t == 0) pn[0] = sqrtf(s[0]);
}

// ---------------------------------------------------------------------------
// 2. y = |feat.p|/||p||, gate = 2*sigmoid(y), and gated feat -> f16 (fused).
//    One wave32 per node; lane handles 8 contiguous floats.
// ---------------------------------------------------------------------------
__global__ void score_gate_kernel(const float* __restrict__ feat,
                                  const float* __restrict__ p,
                                  const float* __restrict__ pn,
                                  float* __restrict__ y,
                                  _Float16* __restrict__ af16, int N) {
    int wid  = blockIdx.x * 8 + (threadIdx.x >> 5);
    int lane = threadIdx.x & 31;
    if (wid >= N) return;
    const float* row = feat + (size_t)wid * FDIM;
    float s = 0.f;
#pragma unroll
    for (int j = 0; j < 8; ++j) s += row[lane * 8 + j] * p[lane * 8 + j];
    // butterfly reduce across wave32 -> every lane holds the full dot
#pragma unroll
    for (int m = 16; m > 0; m >>= 1) s += __shfl_xor(s, m, 32);
    float yy   = fabsf(s) / pn[0];
    float gate = 2.f / (1.f + __expf(-yy));
#pragma unroll
    for (int j = 0; j < 8; ++j)
        af16[(size_t)wid * FDIM + lane * 8 + j] = (_Float16)(gate * row[lane * 8 + j]);
    if (lane == 0) y[wid] = yy;
}

// ---------------------------------------------------------------------------
// 3. W[F][H*O] -> Wt[n][k] f16 (transposed for row-major B fragment loads)
// ---------------------------------------------------------------------------
__global__ void wt_kernel(const float* __restrict__ W, _Float16* __restrict__ Wt) {
    int t = blockIdx.x * blockDim.x + threadIdx.x;   // 65536
    int k = t >> 8, n = t & 255;
    Wt[(size_t)n * FDIM + k] = (_Float16)W[t];
}

// ---------------------------------------------------------------------------
// 4. Hard top-8 neighbor selection. One wave32 per node, 2 slots per lane,
//    8 rounds of max-reduce with lowest-index tie-break (matches top_k order).
// ---------------------------------------------------------------------------
__global__ void topk_kernel(const int* __restrict__ neighbors,
                            const float* __restrict__ y,
                            int* __restrict__ src, int N) {
    int n    = blockIdx.x * 8 + (threadIdx.x >> 5);
    int lane = threadIdx.x & 31;
    if (n >= N) return;
    int   id0 = neighbors[(size_t)n * DNEI + lane];
    int   id1 = neighbors[(size_t)n * DNEI + lane + 32];
    float v0  = y[id0];
    float v1  = y[id1];
#pragma unroll
    for (int k = 0; k < TOPK; ++k) {
        float bv; int bd;
        if (v0 >= v1) { bv = v0; bd = lane; } else { bv = v1; bd = lane + 32; }
#pragma unroll
        for (int m = 16; m > 0; m >>= 1) {
            float ov = __shfl_xor(bv, m, 32);
            int   od = __shfl_xor(bd, m, 32);
            if (ov > bv || (ov == bv && od < bd)) { bv = ov; bd = od; }
        }
        if (lane == 0) src[(size_t)n * TOPK + k] = neighbors[(size_t)n * DNEI + bd];
        if (bd == lane)           v0 = -INFINITY;
        else if (bd == lane + 32) v1 = -INFINITY;
    }
}

// ---------------------------------------------------------------------------
// 5. ft = Af16 @ Wt^T  via v_wmma_f32_16x16x32_f16.
//    One 512-thread block (16 waves) per M-tile: each wave owns one of the 16
//    N-tiles, so all redundant A-fragment reads hit the WGP$.
//    A fragment per ISA layout: lane m = L&15; lanes 0-15 take K {0..7,16..23},
//    lanes 16-31 take K {8..15,24..31}  -> two 16B loads per lane.
// ---------------------------------------------------------------------------
union VH { v16h v; uint4 u[2]; };

__global__ void gemm_wmma_kernel(const _Float16* __restrict__ A,
                                 const _Float16* __restrict__ Bt,
                                 float* __restrict__ C, int N) {
    int mt   = blockIdx.x;                       // M tile (16 rows)
    int nt   = threadIdx.y;                      // N tile (16 cols), 16 per block
    int lane = threadIdx.x;
    int m0   = mt << 4;
    if (m0 >= N) return;
    int sel  = (lane >> 4) & 1;                  // K-half select per lane group
    int mrow = m0 + (lane & 15);
    if (mrow >= N) mrow = N - 1;                 // clamp tail loads
    int ncol = (nt << 4) + (lane & 15);
    const _Float16* arow = A  + (size_t)mrow * FDIM;
    const _Float16* brow = Bt + (size_t)ncol * FDIM;
    v8f acc = {};
#pragma unroll
    for (int kk = 0; kk < FDIM; kk += 32) {
        VH a, b;
        a.u[0] = *(const uint4*)(arow + kk + sel * 8);
        a.u[1] = *(const uint4*)(arow + kk + 16 + sel * 8);
        b.u[0] = *(const uint4*)(brow + kk + sel * 8);
        b.u[1] = *(const uint4*)(brow + kk + 16 + sel * 8);
        acc = __builtin_amdgcn_wmma_f32_16x16x32_f16(
            /*neg_a=*/false, a.v, /*neg_b=*/false, b.v,
            /*c_mod=*/(short)0, acc, /*reuse_a=*/false, /*reuse_b=*/false);
    }
    // C layout: VGPR vg -> row m0 + vg + 8*sel, col = ncol
    float* crow = C + (size_t)m0 * (HEADS * ODIM) + ncol;
    if (m0 + 16 <= N) {
        // full tile: wave-uniform, branch-free stores
#pragma unroll
        for (int vg = 0; vg < 8; ++vg)
            crow[(size_t)(vg + sel * 8) * (HEADS * ODIM)] = acc[vg];
    } else {
#pragma unroll
        for (int vg = 0; vg < 8; ++vg) {
            int r = vg + sel * 8;
            if (m0 + r < N) crow[(size_t)r * (HEADS * ODIM)] = acc[vg];
        }
    }
}

// ---------------------------------------------------------------------------
// 6. el/er: one block per node, one wave per head (h = tid>>5, o = tid&31)
// ---------------------------------------------------------------------------
__global__ void elr_kernel(const float* __restrict__ ft,
                           const float* __restrict__ attn_l,
                           const float* __restrict__ attn_r,
                           float* __restrict__ el, float* __restrict__ er, int N) {
    int n = blockIdx.x, t = threadIdx.x;
    int h = t >> 5, o = t & 31;
    float f  = ft[(size_t)n * FDIM + t];
    float pl = f * attn_l[h * ODIM + o];
    float pr = f * attn_r[h * ODIM + o];
#pragma unroll
    for (int m = 16; m > 0; m >>= 1) {
        pl += __shfl_xor(pl, m, 32);
        pr += __shfl_xor(pr, m, 32);
    }
    if (o == 0) { el[(size_t)n * HEADS + h] = pl; er[(size_t)n * HEADS + h] = pr; }
}

// ---------------------------------------------------------------------------
// 7. edge softmax + aggregation + ELU. Block per node (256 thr: h*32+o).
//    ft (51 MB) sits inside the 192 MB L2 -> the random ft[src] gathers
//    are L2 hits; each wave reads coalesced 128B per src row.
// ---------------------------------------------------------------------------
__global__ void final_kernel(const int* __restrict__ src,
                             const float* __restrict__ el,
                             const float* __restrict__ er,
                             const float* __restrict__ ft,
                             float* __restrict__ out, int N) {
    __shared__ int   s_src[TOPK];
    __shared__ float s_e[TOPK * HEADS];          // leaky-relu'd edge scores [k][h]
    int n = blockIdx.x, t = threadIdx.x;
    if (t < TOPK) s_src[t] = src[(size_t)n * TOPK + t];
    __syncthreads();
    if (t < TOPK) __builtin_prefetch(&ft[(size_t)s_src[t] * FDIM], 0, 1);
    if (t < TOPK * HEADS) {
        int k = t >> 3, h = t & 7;
        float e = el[(size_t)s_src[k] * HEADS + h] + er[(size_t)n * HEADS + h];
        s_e[t] = e > 0.f ? e : NEG_SLOPE * e;
    }
    __syncthreads();
    int h = t >> 5;
    float e[TOPK], mx = -1e30f;
#pragma unroll
    for (int k = 0; k < TOPK; ++k) { e[k] = s_e[k * HEADS + h]; mx = fmaxf(mx, e[k]); }
    float sum = 0.f;
#pragma unroll
    for (int k = 0; k < TOPK; ++k) { e[k] = __expf(e[k] - mx); sum += e[k]; }
    float inv = 1.f / sum;
    float acc = 0.f;
#pragma unroll
    for (int k = 0; k < TOPK; ++k)
        acc += e[k] * inv * ft[(size_t)s_src[k] * FDIM + t];
    out[(size_t)n * FDIM + t] = acc > 0.f ? acc : (__expf(acc) - 1.f);
}

// ---------------------------------------------------------------------------
extern "C" void kernel_launch(void* const* d_in, const int* in_sizes, int n_in,
                              void* d_out, int out_size, void* d_ws, size_t ws_size,
                              hipStream_t stream) {
    const float* feat      = (const float*)d_in[0];
    const int*   neighbors = (const int*)d_in[1];
    const float* p         = (const float*)d_in[2];
    const float* W         = (const float*)d_in[3];
    const float* attn_l    = (const float*)d_in[4];
    const float* attn_r    = (const float*)d_in[5];
    const int N = in_sizes[0] / FDIM;

    char* ws = (char*)d_ws;
    auto alignup = [](size_t x) { return (x + 255) & ~(size_t)255; };
    size_t off = 0;
    float*     pn   = (float*)(ws + off);     off = alignup(off + 4);
    float*     y    = (float*)(ws + off);     off = alignup(off + (size_t)N * 4);
    _Float16*  af16 = (_Float16*)(ws + off);  off = alignup(off + (size_t)N * FDIM * 2);
    _Float16*  Wt   = (_Float16*)(ws + off);  off = alignup(off + (size_t)FDIM * FDIM * 2);
    float*     ft   = (float*)(ws + off);     off = alignup(off + (size_t)N * FDIM * 4);
    float*     el   = (float*)(ws + off);     off = alignup(off + (size_t)N * HEADS * 4);
    float*     er   = (float*)(ws + off);     off = alignup(off + (size_t)N * HEADS * 4);
    int*       src  = (int*)(ws + off);       off = alignup(off + (size_t)N * TOPK * 4);
    (void)ws_size; (void)n_in; (void)out_size;

    pnorm_kernel<<<1, FDIM, 0, stream>>>(p, pn);
    score_gate_kernel<<<(N + 7) / 8, 256, 0, stream>>>(feat, p, pn, y, af16, N);
    wt_kernel<<<FDIM, 256, 0, stream>>>(W, Wt);
    topk_kernel<<<(N + 7) / 8, 256, 0, stream>>>(neighbors, y, src, N);

    dim3 ggrid((N + 15) / 16);      // one block per M tile
    dim3 gblk(32, 16);              // 16 waves = all 16 N-tiles share A in WGP$
    gemm_wmma_kernel<<<ggrid, gblk, 0, stream>>>(af16, Wt, ft, N);

    elr_kernel<<<N, 256, 0, stream>>>(ft, attn_l, attn_r, el, er, N);
    final_kernel<<<N, 256, 0, stream>>>(src, el, er, ft, (float*)d_out, N);
}
